// GAT_14302241095714
// MI455X (gfx1250) — compile-verified, matrix operand order checked by reading, not run on hardware
//
#include <hip/hip_runtime.h>
#include <hip/hip_bf16.h>
#include <math.h>

typedef __attribute__((ext_vector_type(16))) _Float16 v16h;
typedef __attribute__((ext_vector_type(8)))  _Float16 v8h;
typedef __attribute__((ext_vector_type(8)))  float    v8f;

#define NN        50000
#define EE        800000
#define INDIM     128
#define HIDC      64
#define NHEADS    4
#define OUTDIM    64
#define NGRAPHS   128
#define NEGSLOPE  0.2f
#define ETOT      (EE + NN)

// ---------------- utility kernels ----------------

__global__ void fill_f32_kernel(float* __restrict__ p, float v, long long n) {
  long long i = (long long)blockIdx.x * blockDim.x + threadIdx.x;
  if (i < n) p[i] = v;
}

__global__ void cvt_f16_kernel(const float* __restrict__ in, _Float16* __restrict__ out, long long n) {
  long long i = (long long)blockIdx.x * blockDim.x + threadIdx.x;
  if (i < n) out[i] = (_Float16)in[i];
}

// in: [rows, cols] row-major f32  ->  out: [cols, rows] row-major f16 (i.e. transpose)
__global__ void cvt_f16_T_kernel(const float* __restrict__ in, _Float16* __restrict__ out,
                                 int rows, int cols) {
  long long i = (long long)blockIdx.x * blockDim.x + threadIdx.x;
  long long tot = (long long)rows * cols;
  if (i >= tot) return;
  int r = (int)(i / cols);
  int c = (int)(i - (long long)r * cols);
  out[(long long)c * rows + r] = (_Float16)in[i];
}

// ---------------- WMMA GEMM: C[M,Ncols] = A[M,K] * B[K,Ncols], Bt = B^T [Ncols,K] ----------------
// One wave per 16x16 output tile. K multiple of 32, all dims multiples of 16.
__global__ __launch_bounds__(32) void wmma_gemm_kernel(const _Float16* __restrict__ A,
                                                       const _Float16* __restrict__ Bt,
                                                       float* __restrict__ C,
                                                       int K, int Ncols) {
  const int row0 = blockIdx.x << 4;
  const int col0 = blockIdx.y << 4;
  const int lane = threadIdx.x;          // 0..31, full wave
  const int r    = lane & 15;
  const int hs   = (lane >> 4) << 3;     // 0 or 8 (K-chunk select per half-wave)

  const _Float16* a_ptr = A  + (long long)(row0 + r) * K + hs;
  const _Float16* b_ptr = Bt + (long long)(col0 + r) * K + hs;

  v8f acc = {};
  for (int k0 = 0; k0 < K; k0 += 32) {
    v16h av, bv;
    // 16-bit A-matrix 16x32 layout: lanes 0-15 K={0..7,16..23}, lanes 16-31 K={8..15,24..31}
    ((v8h*)&av)[0] = *(const v8h*)(a_ptr + k0);
    ((v8h*)&av)[1] = *(const v8h*)(a_ptr + k0 + 16);
    ((v8h*)&bv)[0] = *(const v8h*)(b_ptr + k0);
    ((v8h*)&bv)[1] = *(const v8h*)(b_ptr + k0 + 16);
    acc = __builtin_amdgcn_wmma_f32_16x16x32_f16(false, av, false, bv,
                                                 (short)0, acc, false, false);
  }

  // C/D layout: lane holds column n = lane&15; VGPR j holds row m = j + (lane>=16 ? 8 : 0)
  float* c_ptr = C + (long long)row0 * Ncols + col0 + r;
  const int mbase = (lane >> 4) << 3;
#pragma unroll
  for (int j = 0; j < 8; ++j)
    c_ptr[(long long)(mbase + j) * Ncols] = acc[j];
}

// ---------------- attention node scores: a_s[n,h] = sum_c H[n,h,c]*att_s[h,c] ----------------
__global__ void att_scores_kernel(const float* __restrict__ H,
                                  const float* __restrict__ atts,
                                  const float* __restrict__ attd,
                                  float* __restrict__ as_, float* __restrict__ ad_,
                                  int Hh, int C) {
  __shared__ float ss[256];
  __shared__ float sd[256];
  const int n = blockIdx.x;
  const int t = threadIdx.x;            // t < Hh*C (<= 256)
  const int F = Hh * C;
  float v = H[(long long)n * F + t];
  ss[t] = v * atts[t];
  sd[t] = v * attd[t];
  __syncthreads();
  for (int off = C >> 1; off > 0; off >>= 1) {
    if ((t & (C - 1)) < off) { ss[t] += ss[t + off]; sd[t] += sd[t + off]; }
    __syncthreads();
  }
  if ((t & (C - 1)) == 0) {
    int h = t / C;
    as_[(long long)n * Hh + h] = ss[t];
    ad_[(long long)n * Hh + h] = sd[t];
  }
}

// ---------------- float atomic max (signed/unsigned int ordering trick) ----------------
__device__ __forceinline__ void atomicMaxF(float* addr, float val) {
  if (val >= 0.0f) atomicMax((int*)addr, __float_as_int(val));
  else             atomicMin((unsigned int*)addr, __float_as_uint(val));
}

// ---------------- edge pass A: alpha = leakyrelu(a_s[src]+a_d[dst]); segment max ----------------
__global__ void edge_alpha_kernel(const int* __restrict__ ei,
                                  const float* __restrict__ as_, const float* __restrict__ ad_,
                                  float* __restrict__ alpha, float* __restrict__ amax, int Hh) {
  long long tid = (long long)blockIdx.x * blockDim.x + threadIdx.x;
  long long total = (long long)ETOT * Hh;
  if (tid >= total) return;
  int e = (int)(tid / Hh);
  int h = (int)(tid - (long long)e * Hh);
  int src, dst;
  if (e < EE) { src = ei[e]; dst = ei[EE + e]; }
  else        { src = dst = e - EE; }           // self loop
  float a = as_[(long long)src * Hh + h] + ad_[(long long)dst * Hh + h];
  a = a > 0.0f ? a : a * NEGSLOPE;
  alpha[tid] = a;
  atomicMaxF(&amax[(long long)dst * Hh + h], a);
}

// ---------------- edge pass B: ex = exp(alpha - amax[dst]); segment sum; alpha <- ex ----------------
__global__ void edge_exp_kernel(const int* __restrict__ ei,
                                float* __restrict__ alpha,
                                const float* __restrict__ amax,
                                float* __restrict__ denom, int Hh) {
  long long tid = (long long)blockIdx.x * blockDim.x + threadIdx.x;
  long long total = (long long)ETOT * Hh;
  if (tid >= total) return;
  int e = (int)(tid / Hh);
  int h = (int)(tid - (long long)e * Hh);
  int dst;
  if (e < EE) dst = ei[EE + e];
  else        dst = e - EE;
  float ex = expf(alpha[tid] - amax[(long long)dst * Hh + h]);
  alpha[tid] = ex;
  atomicAdd(&denom[(long long)dst * Hh + h], ex);
}

// ---------------- edge pass C: out[dst] += H[src] * ex/denom[dst] ----------------
__global__ void edge_scatter_kernel(const int* __restrict__ ei,
                                    const float* __restrict__ H,
                                    const float* __restrict__ exb,
                                    const float* __restrict__ denom,
                                    float* __restrict__ outacc, int Hh, int C) {
  const int e = blockIdx.x;
  const int t = threadIdx.x;            // t < Hh*C
  const int F = Hh * C;
  int src, dst;
  if (e < EE) { src = ei[e]; dst = ei[EE + e]; }
  else        { src = dst = e - EE; }
  const int h = t / C;
  float coef = exb[(long long)e * Hh + h] / (denom[(long long)dst * Hh + h] + 1e-16f);
  atomicAdd(&outacc[(long long)dst * F + t], H[(long long)src * F + t] * coef);
}

// ---------------- out = relu(out + bias); optional f16 copy ----------------
__global__ void bias_relu_kernel(float* __restrict__ acc, const float* __restrict__ bias,
                                 _Float16* __restrict__ outh, int F, long long total) {
  long long i = (long long)blockIdx.x * blockDim.x + threadIdx.x;
  if (i >= total) return;
  float v = acc[i] + bias[(int)(i % F)];
  v = v > 0.0f ? v : 0.0f;
  acc[i] = v;
  if (outh) outh[i] = (_Float16)v;
}

// ---------------- graph counting + pooling ----------------
__global__ void count_kernel(const int* __restrict__ batch, float* __restrict__ cnt) {
  int n = blockIdx.x * blockDim.x + threadIdx.x;
  if (n < NN) atomicAdd(&cnt[batch[n]], 1.0f);
}

__global__ void pool_kernel(const float* __restrict__ x, const int* __restrict__ batch,
                            float* __restrict__ pooled, int C) {
  int n = blockIdx.x;
  int c = threadIdx.x;
  int g = batch[n];
  atomicAdd(&pooled[(long long)g * C + c], x[(long long)n * C + c]);
}

// ---------------- final classifier: out[g,j] = (pooled[g,:]/cnt[g]) @ Wfc + bfc ----------------
__global__ void fc_kernel(const float* __restrict__ pooled, const float* __restrict__ cnt,
                          const float* __restrict__ Wfc, const float* __restrict__ bfc,
                          float* __restrict__ out) {
  int g = blockIdx.x;   // 128
  int j = threadIdx.x;  // 64
  float inv = 1.0f / fmaxf(cnt[g], 1.0f);
  float acc = bfc[j];
  for (int k = 0; k < HIDC; ++k)
    acc += pooled[(long long)g * HIDC + k] * inv * Wfc[(long long)k * OUTDIM + j];
  out[(long long)g * OUTDIM + j] = acc;
}

// ---------------- host side ----------------

static inline long long cdiv_ll(long long a, long long b) { return (a + b - 1) / b; }

extern "C" void kernel_launch(void* const* d_in, const int* in_sizes, int n_in,
                              void* d_out, int out_size, void* d_ws, size_t ws_size,
                              hipStream_t stream) {
  const float* x     = (const float*)d_in[0];
  const int*   ei    = (const int*)d_in[1];   // [2,E]: row0=src, row1=dst
  const int*   batch = (const int*)d_in[2];
  const float* W1    = (const float*)d_in[3]; // [128,256]
  const float* atts1 = (const float*)d_in[4]; // [4,64]
  const float* attd1 = (const float*)d_in[5];
  const float* b1    = (const float*)d_in[6]; // [256]
  const float* W2    = (const float*)d_in[7]; // [256,64]
  const float* atts2 = (const float*)d_in[8]; // [1,64]
  const float* attd2 = (const float*)d_in[9];
  const float* b2    = (const float*)d_in[10]; // [64]
  const float* Wfc   = (const float*)d_in[11]; // [64,64]
  const float* bfc   = (const float*)d_in[12]; // [64]
  float* out = (float*)d_out;

  // ---- workspace carve-out (256B aligned) ----
  char* ws = (char*)d_ws;
  size_t off = 0;
  auto carve = [&](size_t bytes) -> void* {
    void* p = ws + off;
    off = (off + bytes + 255) & ~(size_t)255;
    return p;
  };
  _Float16* x_h    = (_Float16*)carve((size_t)NN * INDIM * 2);
  _Float16* w1t_h  = (_Float16*)carve((size_t)256 * 128 * 2);        // W1^T [256,128]
  _Float16* w2t_h  = (_Float16*)carve((size_t)64 * 256 * 2);         // W2^T [64,256]
  float*    h1     = (float*)carve((size_t)NN * 256 * 4);            // x @ W1
  float*    as1    = (float*)carve((size_t)NN * NHEADS * 4);
  float*    ad1    = (float*)carve((size_t)NN * NHEADS * 4);
  float*    amax1  = (float*)carve((size_t)NN * NHEADS * 4);
  float*    den1   = (float*)carve((size_t)NN * NHEADS * 4);
  float*    alpha1 = (float*)carve((size_t)ETOT * NHEADS * 4);
  float*    out1   = (float*)carve((size_t)NN * 256 * 4);            // conv1 output (f32)
  _Float16* h1r_h  = (_Float16*)carve((size_t)NN * 256 * 2);         // relu(conv1) f16
  float*    h2     = (float*)carve((size_t)NN * 64 * 4);             // h1r @ W2
  float*    as2    = (float*)carve((size_t)NN * 4);
  float*    ad2    = (float*)carve((size_t)NN * 4);
  float*    amax2  = (float*)carve((size_t)NN * 4);
  float*    den2   = (float*)carve((size_t)NN * 4);
  float*    alpha2 = (float*)carve((size_t)ETOT * 4);
  float*    out2   = (float*)carve((size_t)NN * 64 * 4);             // conv2 output (f32)
  float*    pooled = (float*)carve((size_t)NGRAPHS * 64 * 4);
  float*    cnt    = (float*)carve((size_t)NGRAPHS * 4);

  const float NEG_INF = -__builtin_huge_valf();
  auto fill = [&](float* p, float v, long long n) {
    fill_f32_kernel<<<(int)cdiv_ll(n, 256), 256, 0, stream>>>(p, v, n);
  };

  // ---- stage 0: f16 conversions ----
  {
    long long n = (long long)NN * INDIM;
    cvt_f16_kernel<<<(int)cdiv_ll(n, 256), 256, 0, stream>>>(x, x_h, n);
    cvt_f16_T_kernel<<<(int)cdiv_ll(128LL * 256, 256), 256, 0, stream>>>(W1, w1t_h, 128, 256);
    cvt_f16_T_kernel<<<(int)cdiv_ll(256LL * 64, 256), 256, 0, stream>>>(W2, w2t_h, 256, 64);
  }

  // ---- conv1 ----
  // GEMM1: h1[N,256] = x_h[N,128] @ W1  (WMMA f16 -> f32)
  wmma_gemm_kernel<<<dim3(NN / 16, 256 / 16), 32, 0, stream>>>(x_h, w1t_h, h1, INDIM, 256);
  // node scores
  att_scores_kernel<<<NN, NHEADS * HIDC, 0, stream>>>(h1, atts1, attd1, as1, ad1, NHEADS, HIDC);
  // init segment buffers
  fill(amax1, NEG_INF, (long long)NN * NHEADS);
  fill(den1, 0.0f, (long long)NN * NHEADS);
  fill(out1, 0.0f, (long long)NN * 256);
  // edge passes
  {
    long long tot = (long long)ETOT * NHEADS;
    edge_alpha_kernel<<<(int)cdiv_ll(tot, 256), 256, 0, stream>>>(ei, as1, ad1, alpha1, amax1, NHEADS);
    edge_exp_kernel<<<(int)cdiv_ll(tot, 256), 256, 0, stream>>>(ei, alpha1, amax1, den1, NHEADS);
    edge_scatter_kernel<<<ETOT, NHEADS * HIDC, 0, stream>>>(ei, h1, alpha1, den1, out1, NHEADS, HIDC);
  }
  // bias + relu + f16 copy for next gemm
  bias_relu_kernel<<<(int)cdiv_ll((long long)NN * 256, 256), 256, 0, stream>>>(
      out1, b1, h1r_h, 256, (long long)NN * 256);

  // ---- conv2 ----
  // GEMM2: h2[N,64] = h1r[N,256] @ W2
  wmma_gemm_kernel<<<dim3(NN / 16, 64 / 16), 32, 0, stream>>>(h1r_h, w2t_h, h2, 256, 64);
  att_scores_kernel<<<NN, 1 * HIDC, 0, stream>>>(h2, atts2, attd2, as2, ad2, 1, HIDC);
  fill(amax2, NEG_INF, NN);
  fill(den2, 0.0f, NN);
  fill(out2, 0.0f, (long long)NN * 64);
  {
    long long tot = (long long)ETOT;
    edge_alpha_kernel<<<(int)cdiv_ll(tot, 256), 256, 0, stream>>>(ei, as2, ad2, alpha2, amax2, 1);
    edge_exp_kernel<<<(int)cdiv_ll(tot, 256), 256, 0, stream>>>(ei, alpha2, amax2, den2, 1);
    edge_scatter_kernel<<<ETOT, 1 * HIDC, 0, stream>>>(ei, h2, alpha2, den2, out2, 1, HIDC);
  }
  bias_relu_kernel<<<(int)cdiv_ll((long long)NN * 64, 256), 256, 0, stream>>>(
      out2, b2, (_Float16*)nullptr, 64, (long long)NN * 64);

  // ---- pooling + classifier ----
  fill(pooled, 0.0f, (long long)NGRAPHS * 64);
  fill(cnt, 0.0f, NGRAPHS);
  count_kernel<<<(int)cdiv_ll(NN, 256), 256, 0, stream>>>(batch, cnt);
  pool_kernel<<<NN, 64, 0, stream>>>(out2, batch, pooled, 64);
  fc_kernel<<<NGRAPHS, OUTDIM, 0, stream>>>(pooled, cnt, Wfc, bfc, out);
}
